// CrossNetMix_7327214207045
// MI455X (gfx1250) — compile-verified, matrix operand order checked by reading, not run on hardware
//
#include <hip/hip_runtime.h>
#include <cstddef>
#include <cstdint>

// ---------------- problem constants ----------------
constexpr int NC = 3;       // cross layers
constexpr int NE = 8;       // experts
constexpr int D  = 1024;    // d_in
constexpr int R  = 64;      // low rank
constexpr int BT = 16384;   // batch
constexpr int BM = 64;      // batch rows per block

// GEMM1: [BM x 1024] @ [1024 x 544]  (cols 0..511 = V per (e,r), 512..519 = Wg^T, rest 0)
constexpr int N1F = 34;     // 544/16 column fragments
constexpr int K1F = 32;     // 1024/32 k fragments
// GEMM2: per expert [BM x 64] @ [64 x 64]
constexpr int N2F = 4;
constexpr int K2F = 2;
// GEMM3: [BM x 512] @ [512 x 1024]
constexpr int N3F = 64;
constexpr int K3F = 16;

constexpr size_t B1_ELEMS_L = (size_t)N1F * K1F * 32 * 16;        // 557056 bf16 / layer
constexpr size_t B3_ELEMS_L = (size_t)N3F * K3F * 32 * 16;        // 524288
constexpr size_t B2_ELEMS_L = (size_t)NE * N2F * K2F * 32 * 16;   // 32768

constexpr int XI_STRIDE = D + 8;     // 1032 bf16 elems (16B-aligned rows, bank-skewed)
constexpr int H_STRIDE  = 512 + 8;   // 520

typedef __attribute__((ext_vector_type(16))) __bf16 v16bf;
typedef __attribute__((ext_vector_type(8)))  float  v8f;

union Frag { v16bf v; uint4 u[2]; };

__device__ __forceinline__ float fast_tanh(float x) {
  x = fminf(10.0f, fmaxf(-10.0f, x));
  float e2 = __expf(2.0f * x);
  return __fdividef(e2 - 1.0f, e2 + 1.0f);
}

__device__ __forceinline__ v8f wmma_bf16(const Frag& a, const Frag& b, v8f c) {
  return __builtin_amdgcn_wmma_f32_16x16x32_bf16(false, a.v, false, b.v,
                                                 (short)0, c, false, false);
}

// ---------------------------------------------------------------------------
// Prep kernels: pack weights into WMMA B-fragment layout, f32 -> bf16.
// B-frag layout per 16x16x32 step: 32 lanes x 16 bf16 contiguous per lane.
// Lane l supplies column n = l%16, k = (l/16)*8 + j (j<8) / +16+(j-8).
// ---------------------------------------------------------------------------
__global__ void prep_b1(const float* __restrict__ V, const float* __restrict__ Wg,
                        __bf16* __restrict__ out) {
  int idx = blockIdx.x * 256 + threadIdx.x;             // 3*34*32*32 threads
  int lane = idx & 31;
  int t = idx >> 5;
  int kf = t & 31; t >>= 5;
  int nf = t % N1F; int layer = t / N1F;
  int n = nf * 16 + (lane & 15);
  int kbase = kf * 32 + (lane >> 4) * 8;
  Frag f;
#pragma unroll
  for (int j = 0; j < 16; ++j) {
    int k = kbase + (j < 8 ? j : j + 8);
    float val = 0.0f;
    if (n < 512) {
      int e = n >> 6, r = n & 63;
      val = V[(((size_t)layer * NE + e) * D + k) * R + r];
    } else if (n < 520) {
      val = Wg[(size_t)(n - 512) * D + k];
    }
    f.v[j] = (__bf16)val;
  }
  uint4* dst = (uint4*)(out + (size_t)idx * 16);
  dst[0] = f.u[0]; dst[1] = f.u[1];
}

__global__ void prep_b3(const float* __restrict__ U, __bf16* __restrict__ out) {
  int idx = blockIdx.x * 256 + threadIdx.x;             // 3*64*16*32 threads
  int lane = idx & 31;
  int t = idx >> 5;
  int kf = t & 15; t >>= 4;
  int nf = t % N3F; int layer = t / N3F;
  int d = nf * 16 + (lane & 15);
  int kbase = kf * 32 + (lane >> 4) * 8;
  Frag f;
#pragma unroll
  for (int j = 0; j < 16; ++j) {
    int kk = kbase + (j < 8 ? j : j + 8);               // kk = e*64 + r
    int e = kk >> 6, r = kk & 63;
    f.v[j] = (__bf16)U[(((size_t)layer * NE + e) * D + d) * R + r];
  }
  uint4* dst = (uint4*)(out + (size_t)idx * 16);
  dst[0] = f.u[0]; dst[1] = f.u[1];
}

__global__ void prep_b2(const float* __restrict__ C, __bf16* __restrict__ out) {
  int idx = blockIdx.x * 256 + threadIdx.x;             // 3*8*4*2*32 threads
  int lane = idx & 31;
  int t = idx >> 5;
  int kf = t & 1; t >>= 1;
  int nf = t & 3; t >>= 2;
  int e = t & 7; int layer = t >> 3;
  int s = nf * 16 + (lane & 15);
  int kbase = kf * 32 + (lane >> 4) * 8;
  Frag f;
#pragma unroll
  for (int j = 0; j < 16; ++j) {
    int r = kbase + (j < 8 ? j : j + 8);
    f.v[j] = (__bf16)C[(((size_t)layer * NE + e) * R + r) * R + s];
  }
  uint4* dst = (uint4*)(out + (size_t)idx * 16);
  dst[0] = f.u[0]; dst[1] = f.u[1];
}

// ---------------------------------------------------------------------------
// One fused cross layer.
// ---------------------------------------------------------------------------
__global__ __launch_bounds__(256) void cross_layer(
    const float* __restrict__ xin, const float* __restrict__ x0,
    float* __restrict__ xout, const float* __restrict__ bias,
    const __bf16* __restrict__ b1, const __bf16* __restrict__ b2,
    const __bf16* __restrict__ b3) {
  extern __shared__ char smem[];
  __bf16* ldsXi = (__bf16*)smem;                                     // 64x1032 bf16
  __bf16* ldsH2 = ldsXi;                                             // reuse as 64x520
  __bf16* ldsH  = (__bf16*)(smem + (size_t)BM * XI_STRIDE * 2);      // 64x520 bf16
  float*  ldsG  = (float*)(smem + (size_t)BM * XI_STRIDE * 2 + (size_t)BM * H_STRIDE * 2);
  float*  ldsW  = ldsG + BM * NE;                                    // [8][64] f32

  const int tid  = threadIdx.x;
  const int lane = tid & 31;
  const int wid  = tid >> 5;
  const int wr   = wid & 3;     // row-fragment owner (4)
  const int wc   = wid >> 2;    // column-pair owner (2)
  const int col  = lane & 15;
  const int half = lane >> 4;
  const int rowblk = blockIdx.x * BM;
  const v8f vzero = {0.f, 0.f, 0.f, 0.f, 0.f, 0.f, 0.f, 0.f};

  // ---- stage xi tile: global f32 -> LDS bf16, row-major ----
  {
    const float4* xin4 = (const float4*)xin + (size_t)rowblk * (D / 4);
#pragma unroll
    for (int it = 0; it < (BM * D / 4) / 256; ++it) {
      int fi = it * 256 + tid;
      int row = fi >> 8;           // 256 float4 per row
      int c4  = fi & 255;
      float4 xv = xin4[fi];
      union { __bf16 h[4]; uint2 u; } p;
      p.h[0] = (__bf16)xv.x; p.h[1] = (__bf16)xv.y;
      p.h[2] = (__bf16)xv.z; p.h[3] = (__bf16)xv.w;
      *(uint2*)&ldsXi[row * XI_STRIDE + c4 * 4] = p.u;
    }
  }
  __syncthreads();

  // ---- GEMM1: h1 = tanh(xi @ B1), cols 0..511 ----
  const __bf16* arowXi = &ldsXi[(wr * 16 + col) * XI_STRIDE + half * 8];
  for (int cc = 0; cc < 8; ++cc) {
    int nf0 = cc * 4 + wc * 2;
    v8f acc0 = vzero, acc1 = vzero;
    const __bf16* bbase = b1 + (size_t)nf0 * K1F * 512 + (size_t)lane * 16;
    if (cc + 1 < 8) __builtin_prefetch(bbase + (size_t)2 * K1F * 512, 0, 1);
#pragma unroll 4
    for (int kf = 0; kf < K1F; ++kf) {
      Frag a, bA, bB;
      const __bf16* ap = arowXi + kf * 32;
      a.u[0] = *(const uint4*)ap;
      a.u[1] = *(const uint4*)(ap + 16);
      const __bf16* bp = bbase + (size_t)kf * 512;
      bA.u[0] = *(const uint4*)bp;
      bA.u[1] = *(const uint4*)(bp + 8);
      const __bf16* bq = bp + (size_t)K1F * 512;
      bB.u[0] = *(const uint4*)bq;
      bB.u[1] = *(const uint4*)(bq + 8);
      acc0 = wmma_bf16(a, bA, acc0);
      acc1 = wmma_bf16(a, bB, acc1);
    }
#pragma unroll
    for (int v = 0; v < 8; ++v) {
      int row = wr * 16 + half * 8 + v;
      ldsH[row * H_STRIDE + nf0 * 16 + col]       = (__bf16)fast_tanh(acc0[v]);
      ldsH[row * H_STRIDE + (nf0 + 1) * 16 + col] = (__bf16)fast_tanh(acc1[v]);
    }
  }
  // ---- gating columns (512..527 hold Wg^T): g = xi @ Wg^T ----
  {
    int nf = 32 + wc;
    v8f acc = vzero;
    const __bf16* bbase = b1 + (size_t)nf * K1F * 512 + (size_t)lane * 16;
#pragma unroll 4
    for (int kf = 0; kf < K1F; ++kf) {
      Frag a, bb;
      const __bf16* ap = arowXi + kf * 32;
      a.u[0] = *(const uint4*)ap;
      a.u[1] = *(const uint4*)(ap + 16);
      const __bf16* bp = bbase + (size_t)kf * 512;
      bb.u[0] = *(const uint4*)bp;
      bb.u[1] = *(const uint4*)(bp + 8);
      acc = wmma_bf16(a, bb, acc);
    }
    if (wc == 0 && col < NE) {
#pragma unroll
      for (int v = 0; v < 8; ++v)
        ldsG[(wr * 16 + half * 8 + v) * NE + col] = acc[v];
    }
  }
  __syncthreads();

  // ---- softmax over experts (one thread per batch row) ----
  if (tid < BM) {
    float g[NE];
    float m = -3.4e38f;
#pragma unroll
    for (int e = 0; e < NE; ++e) { g[e] = ldsG[tid * NE + e]; m = fmaxf(m, g[e]); }
    float s = 0.f;
#pragma unroll
    for (int e = 0; e < NE; ++e) { g[e] = __expf(g[e] - m); s += g[e]; }
    float inv = __fdividef(1.0f, s);
#pragma unroll
    for (int e = 0; e < NE; ++e) ldsW[e * BM + tid] = g[e] * inv;   // expert-major
  }
  __syncthreads();

  // ---- GEMM2 (wave == expert): h2 = w[:,e] * tanh(h1_e @ C_e) ----
  {
    int e = wid;
    for (int nf = 0; nf < N2F; ++nf) {
      v8f acc[4] = {vzero, vzero, vzero, vzero};
#pragma unroll
      for (int kf = 0; kf < K2F; ++kf) {
        Frag bb;
        const __bf16* bp = b2 + ((size_t)((e * N2F + nf) * K2F + kf) * 32 + lane) * 16;
        bb.u[0] = *(const uint4*)bp;
        bb.u[1] = *(const uint4*)(bp + 8);
#pragma unroll
        for (int mr = 0; mr < 4; ++mr) {
          Frag a;
          const __bf16* ap = &ldsH[(mr * 16 + col) * H_STRIDE + e * 64 + kf * 32 + half * 8];
          a.u[0] = *(const uint4*)ap;
          a.u[1] = *(const uint4*)(ap + 16);
          acc[mr] = wmma_bf16(a, bb, acc[mr]);
        }
      }
#pragma unroll
      for (int mr = 0; mr < 4; ++mr) {
        int rbase = mr * 16 + half * 8;
        const float* wp = &ldsW[e * BM + rbase];
        float4 wA = *(const float4*)wp;
        float4 wB = *(const float4*)(wp + 4);
        float wv[8] = {wA.x, wA.y, wA.z, wA.w, wB.x, wB.y, wB.z, wB.w};
        int cbase = e * 64 + nf * 16 + col;
#pragma unroll
        for (int v = 0; v < 8; ++v)
          ldsH2[(rbase + v) * H_STRIDE + cbase] = (__bf16)(fast_tanh(acc[mr][v]) * wv[v]);
      }
    }
  }
  __syncthreads();

  // ---- GEMM3 + fused epilogue: out = xi + x0*(bias + h2 @ B3) ----
  const __bf16* arowH2 = &ldsH2[(wr * 16 + col) * H_STRIDE + half * 8];
  for (int cc = 0; cc < 16; ++cc) {
    int nf0 = cc * 4 + wc * 2;
    v8f acc0 = vzero, acc1 = vzero;
    const __bf16* bbase = b3 + (size_t)nf0 * K3F * 512 + (size_t)lane * 16;
    if (cc + 1 < 16) __builtin_prefetch(bbase + (size_t)2 * K3F * 512, 0, 1);
#pragma unroll 4
    for (int kf = 0; kf < K3F; ++kf) {
      Frag a, bA, bB;
      const __bf16* ap = arowH2 + kf * 32;
      a.u[0] = *(const uint4*)ap;
      a.u[1] = *(const uint4*)(ap + 16);
      const __bf16* bp = bbase + (size_t)kf * 512;
      bA.u[0] = *(const uint4*)bp;
      bA.u[1] = *(const uint4*)(bp + 8);
      const __bf16* bq = bp + (size_t)K3F * 512;
      bB.u[0] = *(const uint4*)bq;
      bB.u[1] = *(const uint4*)(bq + 8);
      acc0 = wmma_bf16(a, bA, acc0);
      acc1 = wmma_bf16(a, bB, acc1);
    }
    int c0 = nf0 * 16 + col;
    int c1 = c0 + 16;
    float bi0 = bias[c0];
    float bi1 = bias[c1];
    int rb = rowblk + wr * 16 + half * 8;
#pragma unroll
    for (int v = 0; v < 8; ++v) {
      size_t o0 = (size_t)(rb + v) * D + c0;
      size_t o1 = (size_t)(rb + v) * D + c1;
      xout[o0] = xin[o0] + x0[o0] * (bi0 + acc0[v]);
      xout[o1] = xin[o1] + x0[o1] * (bi1 + acc1[v]);
    }
  }
}

// ---------------------------------------------------------------------------
extern "C" void kernel_launch(void* const* d_in, const int* in_sizes, int n_in,
                              void* d_out, int out_size, void* d_ws, size_t ws_size,
                              hipStream_t stream) {
  (void)in_sizes; (void)n_in; (void)out_size; (void)ws_size;
  const float* x  = (const float*)d_in[0];
  const float* U  = (const float*)d_in[1];
  const float* V  = (const float*)d_in[2];
  const float* C  = (const float*)d_in[3];
  const float* Wg = (const float*)d_in[4];
  const float* bv = (const float*)d_in[5];
  float* out = (float*)d_out;

  // workspace layout (bytes)
  char* ws = (char*)d_ws;
  __bf16* B1 = (__bf16*)ws;                                    // 3*557056*2 = 3342336
  __bf16* B3 = (__bf16*)(ws + 3342336);                        // 3*524288*2 = 3145728
  __bf16* B2 = (__bf16*)(ws + 6488064);                        // 3*32768*2  = 196608
  float*  xi_mid = (float*)(ws + 6684672);                     // 16384*1024*4

  prep_b1<<<(NC * N1F * K1F * 32) / 256, 256, 0, stream>>>(V, Wg, B1);
  prep_b3<<<(NC * N3F * K3F * 32) / 256, 256, 0, stream>>>(U, B3);
  prep_b2<<<(NC * NE * N2F * K2F * 32) / 256, 256, 0, stream>>>(C, B2);

  const size_t SMEM = (size_t)BM * XI_STRIDE * 2   // xi tile (reused as h2)
                    + (size_t)BM * H_STRIDE * 2    // h1 tile
                    + (size_t)BM * NE * 4          // gating logits
                    + (size_t)NE * BM * 4;         // softmax weights

  for (int i = 0; i < NC; ++i) {
    const float* xin = (i == 0) ? x : ((i == 1) ? (const float*)out : (const float*)xi_mid);
    float* xo = (i == 1) ? xi_mid : out;
    cross_layer<<<BT / BM, 256, SMEM, stream>>>(
        xin, x, xo, bv + (size_t)i * D,
        B1 + (size_t)i * B1_ELEMS_L,
        B2 + (size_t)i * B2_ELEMS_L,
        B3 + (size_t)i * B3_ELEMS_L);
  }
}
